// GATInnerLayer_v2_55216099557463
// MI455X (gfx1250) — matrix operand order repaired and checked by reading, not compiled, not run on hardware
//
#include <hip/hip_runtime.h>
#include <hip/hip_bf16.h>

typedef __attribute__((ext_vector_type(16))) __bf16 v16bf;
typedef __attribute__((ext_vector_type(8)))  __bf16 v8bf;
typedef __attribute__((ext_vector_type(4)))  __bf16 v4bf;
typedef __attribute__((ext_vector_type(2)))  __bf16 v2bf;
typedef __attribute__((ext_vector_type(8)))  float  v8f;

#define NROWS 8192
#define D     128
#define C3    384
#define HDIM  2029

// Padded per-unit K (multiples of 32): 100->128, 342->352, 1587->1600
#define KP0 128
#define KP1 352
#define KP2 1600

// Packed B sizes (elements): per unit = 24 ctiles * ksteps * 512
#define U0_SZ (24 * 4 * 512)
#define U1_SZ (24 * 11 * 512)
#define U2_SZ (24 * 50 * 512)
#define BPACK_SZ (U0_SZ + U1_SZ + U2_SZ)      // 798720
#define BLN_SZ (8 * 12 * 512)                  // 49152

#define WMMA_BF16(a, b, c) \
    __builtin_amdgcn_wmma_f32_16x16x32_bf16(false, (a), false, (b), (short)0, (c), false, false)

static __device__ __forceinline__ v16bf load_afrag(const __bf16* p) {
    // A fragment (16x32 bf16): lane m|half needs K = [k0, k0+8) and [k0+16, k0+24),
    // both 16B-aligned contiguous runs in row-major memory.
    v8bf lo = *(const v8bf*)(p);
    v8bf hi = *(const v8bf*)(p + 16);
    return __builtin_shufflevector(lo, hi, 0,1,2,3,4,5,6,7,8,9,10,11,12,13,14,15);
}

// ---------------------------------------------------------------------------
// Kernel 0: convert h f32 -> three zero-padded row-major bf16 slice buffers.
// hBall = [u0: 8192 x 128][u1: 8192 x 352][u2: 8192 x 1600]
// ---------------------------------------------------------------------------
__global__ __launch_bounds__(256)
void pack_h(const float* __restrict__ h, __bf16* __restrict__ hBall) {
    const int P0 = NROWS * (KP0 / 2);
    const int P1 = NROWS * (KP1 / 2);
    const int P2 = NROWS * (KP2 / 2);
    int tid = blockIdx.x * 256 + threadIdx.x;
    if (tid >= P0 + P1 + P2) return;

    int row, k, F, fOff, KP;
    size_t base;
    if (tid < P0) {
        row = tid / (KP0 / 2); k = (tid - row * (KP0 / 2)) * 2;
        F = 100; fOff = 0; KP = KP0; base = 0;
    } else if (tid < P0 + P1) {
        int rem = tid - P0;
        row = rem / (KP1 / 2); k = (rem - row * (KP1 / 2)) * 2;
        F = 342; fOff = 100; KP = KP1; base = (size_t)NROWS * KP0;
    } else {
        int rem = tid - P0 - P1;
        row = rem / (KP2 / 2); k = (rem - row * (KP2 / 2)) * 2;
        F = 1587; fOff = 442; KP = KP2; base = (size_t)NROWS * (KP0 + KP1);
    }
    const float* hr = h + (size_t)row * HDIM + fOff;
    float v0 = (k     < F) ? hr[k]     : 0.f;
    float v1 = (k + 1 < F) ? hr[k + 1] : 0.f;
    v2bf o; o[0] = (__bf16)v0; o[1] = (__bf16)v1;
    *(v2bf*)(hBall + base + (size_t)row * KP + k) = o;   // 4B-aligned store
}

// ---------------------------------------------------------------------------
// Kernel 1: repack weights (f32 -> bf16) into WMMA B-fragment order.
// B fragment: lane holds col N=lane&15, K = (e>>3)*16 + (lane>>4)*8 + (e&7).
// Bpack[unit][ctile][kstep][lane*16 + e], zero padded where k >= F.
// ---------------------------------------------------------------------------
__global__ __launch_bounds__(256)
void pack_weights(const float* __restrict__ WqT, const float* __restrict__ WkT, const float* __restrict__ WvT,
                  const float* __restrict__ WqA, const float* __restrict__ WkA, const float* __restrict__ WvA,
                  const float* __restrict__ WqV, const float* __restrict__ WkV, const float* __restrict__ WvV,
                  const float* __restrict__ Wln,
                  __bf16* __restrict__ Bpack, __bf16* __restrict__ BpackLn) {
    int tid = blockIdx.x * 256 + threadIdx.x;
    const int B0 = U0_SZ, B1 = U0_SZ + U1_SZ, B2 = BPACK_SZ;
    if (tid >= B2 + BLN_SZ) return;

    if (tid < B2) {
        int base, ks, F;
        const float *Wq, *Wk, *Wv;
        if (tid < B0)      { base = 0;  ks = 4;  F = 100;  Wq = WqT; Wk = WkT; Wv = WvT; }
        else if (tid < B1) { base = B0; ks = 11; F = 342;  Wq = WqA; Wk = WkA; Wv = WvA; }
        else               { base = B1; ks = 50; F = 1587; Wq = WqV; Wk = WkV; Wv = WvV; }
        int rem   = tid - base;
        int perct = ks * 512;
        int ct    = rem / perct;
        int r2    = rem - ct * perct;
        int kstep = r2 >> 9;
        int li    = r2 & 511;
        int lane  = li >> 4;
        int e     = li & 15;
        int c     = ct * 16 + (lane & 15);
        int k     = kstep * 32 + ((e >> 3) << 4) + ((lane >> 4) << 3) + (e & 7);
        float val = 0.f;
        if (k < F) {
            const float* W = (c < 128) ? Wq : (c < 256) ? Wk : Wv;
            val = W[(size_t)(c & 127) * F + k];
        }
        Bpack[tid] = (__bf16)val;
    } else {
        int rem   = tid - B2;
        int perct = 12 * 512;
        int ct    = rem / perct;
        int r2    = rem - ct * perct;
        int kstep = r2 >> 9;
        int li    = r2 & 511;
        int lane  = li >> 4;
        int e     = li & 15;
        int c     = ct * 16 + (lane & 15);
        int k     = kstep * 32 + ((e >> 3) << 4) + ((lane >> 4) << 3) + (e & 7);
        BpackLn[rem] = (__bf16)Wln[(size_t)c * C3 + k];
    }
}

// ---------------------------------------------------------------------------
// Kernel 2: QKV projections, LDS-free. Block = 32 rows x 384 cols; 8 waves.
// Each wave: 2 n-tiles x 3 c-tiles; A fragments loaded directly from the
// padded bf16 slice buffer (global_load_b128 x2 per fragment); B from Bpack.
// qkv[u][n][0:128]=q, [128:256]=k, [256:384]=v  (f32).
// ---------------------------------------------------------------------------
__global__ __launch_bounds__(256)
void qkv_gemm(const __bf16* __restrict__ hBall,
              const __bf16* __restrict__ Bpack,
              float* __restrict__ qkv) {
    const int u = blockIdx.y;
    const __bf16* hB; int KP, ksteps; size_t bbase;
    if (u == 0)      { hB = hBall;                               KP = KP0; ksteps = 4;  bbase = 0; }
    else if (u == 1) { hB = hBall + (size_t)NROWS * KP0;         KP = KP1; ksteps = 11; bbase = U0_SZ; }
    else             { hB = hBall + (size_t)NROWS * (KP0 + KP1); KP = KP2; ksteps = 50; bbase = U0_SZ + U1_SZ; }

    const int t = threadIdx.x, w = t >> 5, lane = t & 31;
    const int m = lane & 15, half = lane >> 4;
    const int n0 = blockIdx.x * 32;

    v8f acc[2][3] = {};
    const __bf16* a0p = hB + (size_t)(n0 + m) * KP + half * 8;
    const __bf16* a1p = a0p + (size_t)16 * KP;
    const __bf16* bp  = Bpack + bbase + (size_t)w * ksteps * 512 + lane * 16;
    const size_t bct  = (size_t)8 * ksteps * 512;   // ctile stride (ct = w + ci*8)

    for (int ks = 0; ks < ksteps; ++ks) {
        v16bf a0 = load_afrag(a0p);
        v16bf a1 = load_afrag(a1p);
        a0p += 32; a1p += 32;
        #pragma unroll
        for (int ci = 0; ci < 3; ++ci) {
            v16bf b = *(const v16bf*)(bp + ci * bct);
            acc[0][ci] = WMMA_BF16(a0, b, acc[0][ci]);
            acc[1][ci] = WMMA_BF16(a1, b, acc[1][ci]);
        }
        bp += 512;
    }

    const int nc = lane & 15;
    #pragma unroll
    for (int nt = 0; nt < 2; ++nt) {
        #pragma unroll
        for (int ci = 0; ci < 3; ++ci) {
            int col = (w + ci * 8) * 16 + nc;
            #pragma unroll
            for (int r = 0; r < 8; ++r) {
                int row = n0 + nt * 16 + half * 8 + r;
                qkv[((size_t)u * NROWS + row) * C3 + col] = acc[nt][ci][r];
            }
        }
    }
}

// ---------------------------------------------------------------------------
// Kernel 3: per-(row, unit) rank-1 attention with exact column softmax.
// score[i,j] = s*q_i*k_j ; col max = s*k_j*(k_j>=0 ? qmax : qmin).
// out_i = sum_j (v_j / Z_j) * exp2(q~_i*k_j - m~_j), q~ = s*log2e*q.
// One wave per (n,u); 8 waves per block. Writes bf16 att [n][384].
// ---------------------------------------------------------------------------
__global__ __launch_bounds__(256)
void attn_softmax(const float* __restrict__ qkv, __bf16* __restrict__ attB) {
    __shared__ float sq[8][128];
    __shared__ float sk[8][128];
    __shared__ float sw[8][128];
    const int t = threadIdx.x, w = t >> 5, lane = t & 31;
    const int ru = blockIdx.x * 8 + w;
    const int u  = ru >> 13;
    const int n  = ru & (NROWS - 1);
    const float* base = qkv + ((size_t)u * NROWS + n) * C3;

    const float sl = 0.088388347648318447f * 1.4426950408889634f;  // (1/sqrt(128))*log2(e)

    float qa[4], ka[4], va[4];
    {
        const float4 q4 = ((const float4*)(base      ))[lane];
        const float4 k4 = ((const float4*)(base + 128))[lane];
        const float4 v4 = ((const float4*)(base + 256))[lane];
        qa[0] = sl * q4.x; qa[1] = sl * q4.y; qa[2] = sl * q4.z; qa[3] = sl * q4.w;
        ka[0] = k4.x; ka[1] = k4.y; ka[2] = k4.z; ka[3] = k4.w;
        va[0] = v4.x; va[1] = v4.y; va[2] = v4.z; va[3] = v4.w;
    }
    float qmax = fmaxf(fmaxf(qa[0], qa[1]), fmaxf(qa[2], qa[3]));   // scaled-q max/min
    float qmin = fminf(fminf(qa[0], qa[1]), fminf(qa[2], qa[3]));
    #pragma unroll
    for (int msk = 16; msk >= 1; msk >>= 1) {
        qmax = fmaxf(qmax, __shfl_xor(qmax, msk, 32));
        qmin = fminf(qmin, __shfl_xor(qmin, msk, 32));
    }
    #pragma unroll
    for (int tt = 0; tt < 4; ++tt) {
        sq[w][lane * 4 + tt] = qa[tt];
        sk[w][lane * 4 + tt] = ka[tt];
    }
    __syncthreads();

    #pragma unroll
    for (int tt = 0; tt < 4; ++tt) {             // my 4 columns j
        float kj = ka[tt];
        float mj = kj * ((kj >= 0.f) ? qmax : qmin);
        float Z = 0.f;
        for (int i = 0; i < 128; ++i)
            Z += exp2f(sq[w][i] * kj - mj);      // fma + v_exp_f32
        sw[w][lane * 4 + tt] = va[tt] / Z;
    }
    __syncthreads();

    float acc[4] = {0.f, 0.f, 0.f, 0.f};
    for (int j = 0; j < 128; ++j) {              // reduce over columns
        float kj = sk[w][j];
        float wj = sw[w][j];
        float mj = kj * ((kj >= 0.f) ? qmax : qmin);
        #pragma unroll
        for (int tt = 0; tt < 4; ++tt)
            acc[tt] += wj * exp2f(qa[tt] * kj - mj);
    }
    v4bf o;
    #pragma unroll
    for (int tt = 0; tt < 4; ++tt) o[tt] = (__bf16)acc[tt];
    *(v4bf*)(attB + (size_t)n * C3 + u * D + lane * 4) = o;   // 8B store
}

// ---------------------------------------------------------------------------
// Kernel 4: final linear att[8192,384] @ Wln.T + bln -> out f32 [8192,128].
// LDS-free: block = 32 rows x 128 cols; wave w owns c-tile w, 2 n-tiles.
// attB stride 384 -> all A-fragment loads 16B-aligned.
// ---------------------------------------------------------------------------
__global__ __launch_bounds__(256)
void final_linear(const __bf16* __restrict__ attB,
                  const __bf16* __restrict__ BpackLn,
                  const float* __restrict__ bln,
                  float* __restrict__ out) {
    const int t = threadIdx.x, w = t >> 5, lane = t & 31;
    const int m = lane & 15, half = lane >> 4;
    const int n0 = blockIdx.x * 32;

    v8f acc[2] = {};
    const __bf16* a0p = attB + (size_t)(n0 + m) * C3 + half * 8;
    const __bf16* a1p = a0p + (size_t)16 * C3;
    const __bf16* bp  = BpackLn + (size_t)w * 12 * 512 + lane * 16;

    for (int ks = 0; ks < 12; ++ks) {
        v16bf a0 = load_afrag(a0p);
        v16bf a1 = load_afrag(a1p);
        v16bf b  = *(const v16bf*)bp;
        acc[0] = WMMA_BF16(a0, b, acc[0]);
        acc[1] = WMMA_BF16(a1, b, acc[1]);
        a0p += 32; a1p += 32; bp += 512;
    }

    const int nc = lane & 15;
    const float bb = bln[w * 16 + nc];
    #pragma unroll
    for (int nt = 0; nt < 2; ++nt)
        #pragma unroll
        for (int r = 0; r < 8; ++r)
            out[(size_t)(n0 + nt * 16 + half * 8 + r) * D + w * 16 + nc] = acc[nt][r] + bb;
}

// ---------------------------------------------------------------------------
extern "C" void kernel_launch(void* const* d_in, const int* in_sizes, int n_in,
                              void* d_out, int out_size, void* d_ws, size_t ws_size,
                              hipStream_t stream) {
    (void)in_sizes; (void)n_in; (void)out_size; (void)ws_size;
    // dict order: g, h, WqT, WkT, WvT, WqA, WkA, WvA, WqV, WkV, WvV, Wln, bln
    const float* h   = (const float*)d_in[1];
    const float* WqT = (const float*)d_in[2];
    const float* WkT = (const float*)d_in[3];
    const float* WvT = (const float*)d_in[4];
    const float* WqA = (const float*)d_in[5];
    const float* WkA = (const float*)d_in[6];
    const float* WvA = (const float*)d_in[7];
    const float* WqV = (const float*)d_in[8];
    const float* WkV = (const float*)d_in[9];
    const float* WvV = (const float*)d_in[10];
    const float* Wln = (const float*)d_in[11];
    const float* bln = (const float*)d_in[12];
    float* out = (float*)d_out;

    char* ws = (char*)d_ws;
    float*  qkv     = (float*)(ws);                 // 3*8192*384*4 = 37748736 B
    __bf16* attB    = (__bf16*)(ws + 37748736);     // 8192*384*2   =  6291456 B
    __bf16* Bpack   = (__bf16*)(ws + 44040192);     // 798720*2     =  1597440 B
    __bf16* BpackLn = (__bf16*)(ws + 45637632);     // 49152*2      =    98304 B
    __bf16* hBall   = (__bf16*)(ws + 45735936);     // 8192*2080*2  = 34078720 B

    const int hTot = NROWS * ((KP0 + KP1 + KP2) / 2);          // 8519680
    pack_h<<<(hTot + 255) / 256, 256, 0, stream>>>(h, hBall);

    const int prepTot = BPACK_SZ + BLN_SZ;                     // 847872
    pack_weights<<<(prepTot + 255) / 256, 256, 0, stream>>>(
        WqT, WkT, WvT, WqA, WkA, WvA, WqV, WkV, WvV, Wln, Bpack, BpackLn);

    qkv_gemm<<<dim3(NROWS / 32, 3), 256, 0, stream>>>(hBall, Bpack, qkv);

    attn_softmax<<<(3 * NROWS) / 8, 256, 0, stream>>>(qkv, attB);

    final_linear<<<NROWS / 32, 256, 0, stream>>>(attB, BpackLn, bln, out);
}
